// SE3_Dynamics_24996709663343
// MI455X (gfx1250) — matrix operand order repaired and checked by reading, not compiled, not run on hardware
//
#include <hip/hip_runtime.h>
#include <math.h>

typedef __attribute__((ext_vector_type(16))) _Float16 v16h;
typedef __attribute__((ext_vector_type(8)))  float    v8f;

#define HID 128
#define NB  32768
#define WPB 4            // waves per block
#define LDH (HID + 8)    // padded f16 LDS row stride
#define PI_F 3.14159265358979323846f

// packed-fragment tile table (per chart): tiles of 32 lanes x 16 f16
//  [0,8)    bw1   (nt)            [8,40)   bw2  (nt*4+kc)
//  [40,52)  bw3   (nt*4+kc)       [52,60)  vw1  (nt)
//  [60,92)  vw2   (nt*4+kc)       [92,96)  vw3  (kc, col0 only, B-layout)
//  [96,128) vw2^T (nt*4+kc)       [128,132) vw1^T (kc, cols 0..5)
//  [132,136) vw3 A-layout (kc)  -- for G2 = (1-h2^2)*w3
#define NTILE 136
#define CHART_ELEMS (NTILE * 512)   // f16 elements per chart

__device__ __forceinline__ v8f wmma16(v16h a, v16h b, v8f c) {
  return __builtin_amdgcn_wmma_f32_16x16x32_f16(false, a, false, b, (short)0, c, false, false);
}

__device__ __forceinline__ v16h load_frag(const _Float16* __restrict__ wp, int tix, int ln) {
  return *reinterpret_cast<const v16h*>(wp + (((size_t)tix * 32 + ln) << 4));
}

__device__ __forceinline__ float clampf(float v, float lo, float hi) {
  return fminf(fmaxf(v, lo), hi);
}

// hardware tanh on CDNA5 (V_TANH_F32); cheap fallback otherwise
__device__ __forceinline__ float fast_tanh(float x) {
#if __has_builtin(__builtin_amdgcn_tanhf)
  return __builtin_amdgcn_tanhf(x);
#elif __has_builtin(__builtin_amdgcn_tanh_f32)
  return __builtin_amdgcn_tanh_f32(x);
#else
  float e = __expf(2.f * x);
  return 1.f - 2.f / (e + 1.f);
#endif
}

// ---------- 6x6 helpers ----------
__device__ __forceinline__ void mm6(const float A[6][6], const float Bm[6][6], float C[6][6]) {
#pragma unroll
  for (int i = 0; i < 6; i++)
#pragma unroll
    for (int j = 0; j < 6; j++) {
      float s = 0.f;
#pragma unroll
      for (int k = 0; k < 6; k++) s += A[i][k] * Bm[k][j];
      C[i][j] = s;
    }
}

__device__ __forceinline__ void inv6(float M[6][12]) {
#pragma unroll
  for (int c = 0; c < 6; c++) {
    float piv = 1.f / M[c][c];
#pragma unroll
    for (int j = 0; j < 12; j++) M[c][j] *= piv;
#pragma unroll
    for (int r = 0; r < 6; r++) {
      if (r == c) continue;
      float f = M[r][c];
#pragma unroll
      for (int j = 0; j < 12; j++) M[r][j] -= f * M[c][j];
    }
  }
}

// k2_inv(q) = inv(I - A/2 + A^2/6 - A^3/24 + A^4/120), A = ad6(q)
__device__ __forceinline__ void k2inv(const float q[6], float X[6][6]) {
  float A[6][6];
#pragma unroll
  for (int i = 0; i < 6; i++)
#pragma unroll
    for (int j = 0; j < 6; j++) A[i][j] = 0.f;
  float w0 = q[0], w1 = q[1], w2 = q[2], v0 = q[3], v1 = q[4], v2 = q[5];
  A[0][1] = -w2; A[0][2] =  w1; A[1][0] =  w2; A[1][2] = -w0; A[2][0] = -w1; A[2][1] =  w0;
  A[3][4] = -w2; A[3][5] =  w1; A[4][3] =  w2; A[4][5] = -w0; A[5][3] = -w1; A[5][4] =  w0;
  A[3][1] = -v2; A[3][2] =  v1; A[4][0] =  v2; A[4][2] = -v0; A[5][0] = -v1; A[5][1] =  v0;
  float A2[6][6];
  mm6(A, A, A2);
  float M[6][12];
#pragma unroll
  for (int i = 0; i < 6; i++)
#pragma unroll
    for (int j = 0; j < 6; j++) {
      M[i][j] = ((i == j) ? 1.f : 0.f) - 0.5f * A[i][j] + A2[i][j] * (1.f / 6.f);
      M[i][6 + j] = (i == j) ? 1.f : 0.f;
    }
  {
    float Tm[6][6];
    mm6(A2, A, Tm);  // A^3
#pragma unroll
    for (int i = 0; i < 6; i++)
#pragma unroll
      for (int j = 0; j < 6; j++) M[i][j] -= Tm[i][j] * (1.f / 24.f);
    mm6(A2, A2, Tm);  // A^4
#pragma unroll
    for (int i = 0; i < 6; i++)
#pragma unroll
      for (int j = 0; j < 6; j++) M[i][j] += Tm[i][j] * (1.f / 120.f);
  }
  inv6(M);
#pragma unroll
  for (int i = 0; i < 6; i++)
#pragma unroll
    for (int j = 0; j < 6; j++) X[i][j] = M[i][6 + j];
}

// ---------- WMMA fragment builders ----------
__device__ __forceinline__ void build_afrag(const _Float16 (*buf)[LDH], int m, int hf, v16h a[4]) {
#pragma unroll
  for (int kc = 0; kc < 4; kc++)
#pragma unroll
    for (int e = 0; e < 16; e++) {
      int K = kc * 32 + 16 * hf + e;
      a[kc][e] = buf[m][K];
    }
}

// dense 128->128 layer with tanh (two-phase: all WMMAs, then activation epilogue)
__device__ __forceinline__ void dense128_tanh(const _Float16 (*in)[LDH], _Float16 (*outb)[LDH],
                                              const _Float16* __restrict__ wp, int tbase,
                                              const float* __restrict__ bias,
                                              int m, int hf, int ln) {
  v16h aH[4];
  build_afrag(in, m, hf, aH);
  float bb[8];
#pragma unroll
  for (int nt = 0; nt < 8; nt++) bb[nt] = bias[nt * 16 + m];
  v8f cs[8];
#pragma unroll
  for (int nt = 0; nt < 8; nt++) {
    v8f c = {0.f, 0.f, 0.f, 0.f, 0.f, 0.f, 0.f, 0.f};
#pragma unroll
    for (int kc = 0; kc < 4; kc++)
      c = wmma16(aH[kc], load_frag(wp, tbase + nt * 4 + kc, ln), c);
    cs[nt] = c;
  }
#pragma unroll
  for (int nt = 0; nt < 8; nt++) {
    int col = nt * 16 + m;
#pragma unroll
    for (int r = 0; r < 8; r++)
      outb[r + 8 * hf][col] = (_Float16)fast_tanh(cs[nt][r] + bb[nt]);
  }
}

// =============== weight pre-pack kernel: f32 matrices -> fragment-major f16 ===============
__global__ __launch_bounds__(256) void pack_weights(
    const float* __restrict__ Vw1, const float* __restrict__ Vw2, const float* __restrict__ Vw3,
    const float* __restrict__ Bw1, const float* __restrict__ Bw2, const float* __restrict__ Bw3,
    _Float16* __restrict__ wpack) {
  int idx = blockIdx.x * 256 + threadIdx.x;  // [4][NTILE][32][16]
  if (idx >= 4 * CHART_ELEMS) return;
  int e   = idx & 15;
  int ln  = (idx >> 4) & 31;
  int tix = (idx >> 9) % NTILE;
  int k   = idx / CHART_ELEMS;
  int m  = ln & 15;
  int hf = ln >> 4;
  float val = 0.f;
  if (tix < 8) {                       // bw1 (12 x HID)
    int col = tix * 16 + m;
    int K = 16 * hf + e;
    val = (K < 12) ? Bw1[k * 12 * HID + K * HID + col] : 0.f;
  } else if (tix < 40) {               // bw2 (HID x HID)
    int t = tix - 8, nt = t >> 2, kc = t & 3;
    int col = nt * 16 + m;
    int K = kc * 32 + 16 * hf + e;
    val = Bw2[k * HID * HID + K * HID + col];
  } else if (tix < 52) {               // bw3 (HID x 36)
    int t = tix - 40, nt = t >> 2, kc = t & 3;
    int col = nt * 16 + m;
    int K = kc * 32 + 16 * hf + e;
    val = (col < 36) ? Bw3[k * HID * 36 + K * 36 + col] : 0.f;
  } else if (tix < 60) {               // vw1 (6 x HID)
    int nt = tix - 52;
    int col = nt * 16 + m;
    int K = 16 * hf + e;
    val = (K < 6) ? Vw1[k * 6 * HID + K * HID + col] : 0.f;
  } else if (tix < 92) {               // vw2 (HID x HID)
    int t = tix - 60, nt = t >> 2, kc = t & 3;
    int col = nt * 16 + m;
    int K = kc * 32 + 16 * hf + e;
    val = Vw2[k * HID * HID + K * HID + col];
  } else if (tix < 96) {               // vw3 (HID x 1), B-layout, valid col 0 only
    int kc = tix - 92;
    int K = kc * 32 + 16 * hf + e;
    val = (m == 0) ? Vw3[k * HID + K] : 0.f;
  } else if (tix < 128) {              // vw2^T
    int t = tix - 96, nt = t >> 2, kc = t & 3;
    int col = nt * 16 + m;
    int K = kc * 32 + 16 * hf + e;
    val = Vw2[k * HID * HID + col * HID + K];
  } else if (tix < 132) {              // vw1^T (cols 0..5)
    int kc = tix - 128;
    int K = kc * 32 + 16 * hf + e;
    val = (m < 6) ? Vw1[k * 6 * HID + m * HID + K] : 0.f;
  } else {                             // vw3 in A-layout (for G2 elementwise product)
    int kc = tix - 132;
    int K = kc * 32 + 16 * hf + e;
    val = Vw3[k * HID + K];
  }
  wpack[idx] = (_Float16)val;
}

// =============== main kernel ===============
__global__ __launch_bounds__(32 * WPB, 2) void se3_dyn_kernel(
    const float* __restrict__ x, const int* __restrict__ chart_idx,
    const float* __restrict__ I_diag,
    const float* __restrict__ Vb1, const float* __restrict__ Vb2,
    const float* __restrict__ Vb3,
    const float* __restrict__ Bb1, const float* __restrict__ Bb2,
    const float* __restrict__ Bb3,
    const _Float16* __restrict__ wpack,
    float* __restrict__ out) {
  __shared__ _Float16 zbuf16[WPB][16][64];   // cols 0..31: z (12 valid); 32..63: z_V (6 valid)
  __shared__ _Float16 act1[WPB][16][LDH];
  __shared__ _Float16 act2[WPB][16][LDH];
  __shared__ float    outbm[WPB][16][36];
  __shared__ float    vbuf[WPB][16];
  __shared__ float    gvbuf[WPB][16][6];

  const int ln   = threadIdx.x & 31;
  const int wid  = threadIdx.x >> 5;
  const int m    = ln & 15;
  const int hf   = ln >> 4;
  const int tile = blockIdx.x * WPB + wid;
  const int s    = tile * 16 + m;

  const float dt0[3] = {1.f, 1.f, 1.f};
  const float dt1[3] = {1.f, -1.f, -1.f};
  const float dt2[3] = {-1.f, 1.f, -1.f};
  const float dt3[3] = {-1.f, -1.f, 1.f};

  // persistent per-sample state (valid on lanes hf==0)
  float P[6], T[6], Bsum[6], dVa[6], dqi[6];
  float Rm[3][3], pm[3];
  float mv[4][3], gco[4], w4[4], Ssum = 1.f;
  int ci = 0;
#pragma unroll
  for (int i = 0; i < 6; i++) { Bsum[i] = 0.f; dVa[i] = 0.f; dqi[i] = 0.f; P[i] = 0.f; T[i] = 0.f; }
#pragma unroll
  for (int k = 0; k < 4; k++) { w4[k] = 0.f; gco[k] = 0.f;
#pragma unroll
    for (int i = 0; i < 3; i++) mv[k][i] = 0.f; }
#pragma unroll
  for (int i = 0; i < 3; i++) { pm[i] = 0.f;
#pragma unroll
    for (int j = 0; j < 3; j++) Rm[i][j] = 0.f; }

  if (hf == 0) {
    ci = chart_idx[s];
    float q0[6];
#pragma unroll
    for (int i = 0; i < 6; i++) {
      q0[i] = x[s * 12 + i];
      P[i]  = x[s * 12 + 6 + i];
      T[i]  = P[i] / I_diag[i];
    }
    // ---- exp_se3(q0) ----
    float t2 = q0[0] * q0[0] + q0[1] * q0[1] + q0[2] * q0[2];
    float t  = sqrtf(t2 + 1e-20f);
    bool  sml = t < 1e-4f;
    float ts = sml ? 1.f : t;
    float sn = __sinf(ts), cs = __cosf(ts);
    float ea = sml ? (1.f - t2 / 6.f) : (sn / ts);
    float eb = sml ? (0.5f - t2 / 24.f) : ((1.f - cs) / (ts * ts));
    float ec = sml ? (1.f / 6.f - t2 / 120.f) : ((ts - sn) / (ts * ts * ts));
    float W[3][3] = {{0.f, -q0[2], q0[1]}, {q0[2], 0.f, -q0[0]}, {-q0[1], q0[0], 0.f}};
    float W2[3][3];
#pragma unroll
    for (int i = 0; i < 3; i++)
#pragma unroll
      for (int j = 0; j < 3; j++) {
        float sacc = 0.f;
#pragma unroll
        for (int k = 0; k < 3; k++) sacc += W[i][k] * W[k][j];
        W2[i][j] = sacc;
      }
    float sg[3] = {1.f, 1.f, 1.f};
    if (ci == 1) { sg[0] = dt1[0]; sg[1] = dt1[1]; sg[2] = dt1[2]; }
    if (ci == 2) { sg[0] = dt2[0]; sg[1] = dt2[1]; sg[2] = dt2[2]; }
    if (ci == 3) { sg[0] = dt3[0]; sg[1] = dt3[1]; sg[2] = dt3[2]; }
#pragma unroll
    for (int i = 0; i < 3; i++) {
#pragma unroll
      for (int j = 0; j < 3; j++) {
        float idm = (i == j) ? 1.f : 0.f;
        Rm[i][j] = sg[i] * (idm + ea * W[i][j] + eb * W2[i][j]);
      }
      float vm0 = ((i == 0) ? 1.f : 0.f) + eb * W[i][0] + ec * W2[i][0];
      float vm1 = ((i == 1) ? 1.f : 0.f) + eb * W[i][1] + ec * W2[i][1];
      float vm2 = ((i == 2) ? 1.f : 0.f) + eb * W[i][2] + ec * W2[i][2];
      pm[i] = sg[i] * (vm0 * q0[3] + vm1 * q0[4] + vm2 * q0[5]);
    }
    // ---- POU at Rm + gradient ingredients ----
    Ssum = 0.f;
    float rr4[4];
#pragma unroll
    for (int j = 0; j < 4; j++) {
      const float* dj = (j == 0) ? dt0 : (j == 1) ? dt1 : (j == 2) ? dt2 : dt3;
      float trj = dj[0] * Rm[0][0] + dj[1] * Rm[1][1] + dj[2] * Rm[2][2];
      float craw = 0.5f * (trj - 1.f);
      bool  clip = (craw <= -1.f + 1e-6f) || (craw >= 1.f - 1e-6f);
      float cc = clampf(craw, -1.f + 1e-6f, 1.f - 1e-6f);
      float th = acosf(cc);
      float base = fmaxf(1.f - (th / PI_F) * (th / PI_F), 0.f);
      rr4[j] = base * base * base;
      Ssum += rr4[j];
      float rprime = -6.f * th / (PI_F * PI_F) * base * base;
      mv[j][0] = dj[1] * Rm[1][2] - dj[2] * Rm[2][1];
      mv[j][1] = dj[2] * Rm[2][0] - dj[0] * Rm[0][2];
      mv[j][2] = dj[0] * Rm[0][1] - dj[1] * Rm[1][0];
      gco[j] = clip ? 0.f : rprime * (-0.5f / sqrtf(fmaxf(1.f - cc * cc, 1e-12f)));
    }
#pragma unroll
    for (int j = 0; j < 4; j++) w4[j] = rr4[j] / Ssum;
  }

  // ================= chart loop =================
  for (int k = 0; k < 4; k++) {
    const _Float16* wp  = wpack + (size_t)k * CHART_ELEMS;
    const float*    vb1 = Vb1 + k * HID;
    const float*    vb2 = Vb2 + k * HID;
    const float*    vb3 = Vb3 + k;
    const float*    bb1 = Bb1 + k * HID;
    const float*    bb2 = Bb2 + k * HID;
    const float*    bb3 = Bb3 + k * 36;

    // prefetch next chart's packed fragments (global_prefetch_b8)
    {
      const _Float16* wpn = wpack + (size_t)((k + 1) & 3) * CHART_ELEMS;
#pragma unroll
      for (int pf = 0; pf < 4; pf++)
        __builtin_prefetch(wpn + ((int)threadIdx.x + pf * 128) * 132, 0, 1);
    }

    // ---- per-chart geometry ----
    float qk[6] = {0.f, 0.f, 0.f, 0.f, 0.f, 0.f};
    float dwq[6] = {0.f, 0.f, 0.f, 0.f, 0.f, 0.f};
    float wk = 0.f;
    if (hf == 0) {
#pragma unroll
      for (int kk = 0; kk < 4; kk++)
        if (kk == k) wk = w4[kk];
      const float* dk = (k == 0) ? dt0 : (k == 1) ? dt1 : (k == 2) ? dt2 : dt3;
      float Rk[3][3], pk[3];
#pragma unroll
      for (int r = 0; r < 3; r++) {
        pk[r] = dk[r] * pm[r];
#pragma unroll
        for (int c = 0; c < 3; c++) Rk[r][c] = dk[r] * Rm[r][c];
      }
      float craw = 0.5f * (Rk[0][0] + Rk[1][1] + Rk[2][2] - 1.f);
      float cc = clampf(craw, -1.f + 1e-6f, 1.f - 1e-6f);
      float th = acosf(cc);
      bool sm2 = th < 1e-4f;
      float ts2 = sm2 ? 1.f : th;
      float f = sm2 ? (0.5f + th * th / 12.f) : (0.5f * ts2 / __sinf(ts2));
      float wv[3] = {f * (Rk[2][1] - Rk[1][2]), f * (Rk[0][2] - Rk[2][0]), f * (Rk[1][0] - Rk[0][1])};
      float t2b = wv[0] * wv[0] + wv[1] * wv[1] + wv[2] * wv[2];
      float tb = sqrtf(t2b + 1e-20f);
      bool smb = tb < 1e-4f;
      float tsb = smb ? 1.f : tb;
      float snb = __sinf(tsb), csb = __cosf(tsb);
      float dcf = smb ? (1.f / 12.f) : ((1.f - 0.5f * tsb * snb / (1.f - csb)) / (tsb * tsb));
      float Wk[3][3] = {{0.f, -wv[2], wv[1]}, {wv[2], 0.f, -wv[0]}, {-wv[1], wv[0], 0.f}};
      float Wk2[3][3];
#pragma unroll
      for (int i = 0; i < 3; i++)
#pragma unroll
        for (int j = 0; j < 3; j++) {
          float sacc = 0.f;
#pragma unroll
          for (int q = 0; q < 3; q++) sacc += Wk[i][q] * Wk[q][j];
          Wk2[i][j] = sacc;
        }
#pragma unroll
      for (int i = 0; i < 3; i++) qk[i] = wv[i];
#pragma unroll
      for (int i = 0; i < 3; i++) {
        float sacc = 0.f;
#pragma unroll
        for (int j = 0; j < 3; j++) {
          float vinv = ((i == j) ? 1.f : 0.f) - 0.5f * Wk[i][j] + dcf * Wk2[i][j];
          sacc += vinv * pk[j];
        }
        qk[3 + i] = sacc;
      }
      float b2 = smb ? (0.5f - t2b / 24.f) : ((1.f - csb) / (tsb * tsb));
      float c2 = smb ? (1.f / 6.f - t2b / 120.f) : ((tsb - snb) / (tsb * tsb * tsb));
      float vs[3] = {0.f, 0.f, 0.f};
#pragma unroll
      for (int j = 0; j < 4; j++) {
        int jj = j ^ k;
        float coef = (((j == 0) ? 1.f : 0.f) - wk) / Ssum * gco[jj];
#pragma unroll
        for (int i = 0; i < 3; i++) vs[i] += coef * mv[jj][i];
      }
#pragma unroll
      for (int i = 0; i < 3; i++) {
        float jri0 = ((0 == i) ? 1.f : 0.f) - b2 * Wk[0][i] + c2 * Wk2[0][i];
        float jri1 = ((1 == i) ? 1.f : 0.f) - b2 * Wk[1][i] + c2 * Wk2[1][i];
        float jri2 = ((2 == i) ? 1.f : 0.f) - b2 * Wk[2][i] + c2 * Wk2[2][i];
        dwq[i] = jri0 * vs[0] + jri1 * vs[1] + jri2 * vs[2];  // (J_r^T vs)_i
        dwq[3 + i] = 0.f;
      }
      // stage z rows pre-converted to f16, zero-padded (vectorized b128 stores)
      v16h z0, z1, zv0;
#pragma unroll
      for (int e = 0; e < 16; e++) {
        float zz = (e < 6) ? qk[e] : ((e < 12) ? P[e - 6] : 0.f);
        z0[e]  = (_Float16)zz;
        z1[e]  = (_Float16)0.f;
        zv0[e] = (_Float16)((e < 6) ? qk[e] : 0.f);
      }
      *reinterpret_cast<v16h*>(&zbuf16[wid][m][0])  = z0;
      *reinterpret_cast<v16h*>(&zbuf16[wid][m][16]) = z1;
      *reinterpret_cast<v16h*>(&zbuf16[wid][m][32]) = zv0;
      *reinterpret_cast<v16h*>(&zbuf16[wid][m][48]) = z1;
    }
    __syncthreads();

    // A fragments for z (16x12) and z_V (16x6): contiguous padded f16 reads
    v16h aZ, aZv;
#pragma unroll
    for (int e = 0; e < 16; e++) {
      int K = 16 * hf + e;
      aZ[e]  = zbuf16[wid][m][K];
      aZv[e] = zbuf16[wid][m][32 + K];
    }

    // ---------- B-MLP forward (two-phase) ----------
    {
      float bb[8];
#pragma unroll
      for (int nt = 0; nt < 8; nt++) bb[nt] = bb1[nt * 16 + m];
      v8f cs[8];
#pragma unroll
      for (int nt = 0; nt < 8; nt++) {
        v8f c = {0.f, 0.f, 0.f, 0.f, 0.f, 0.f, 0.f, 0.f};
        cs[nt] = wmma16(aZ, load_frag(wp, 0 + nt, ln), c);
      }
#pragma unroll
      for (int nt = 0; nt < 8; nt++) {
        int col = nt * 16 + m;
#pragma unroll
        for (int r = 0; r < 8; r++)
          act1[wid][r + 8 * hf][col] = (_Float16)fast_tanh(cs[nt][r] + bb[nt]);
      }
    }
    __syncthreads();
    dense128_tanh(act1[wid], act2[wid], wp, 8, bb2, m, hf, ln);
    __syncthreads();
    {  // layer 3: 16x36 output (two-phase)
      v16h aH2[4];
      build_afrag(act2[wid], m, hf, aH2);
      v8f cs[3];
#pragma unroll
      for (int nt = 0; nt < 3; nt++) {
        v8f c = {0.f, 0.f, 0.f, 0.f, 0.f, 0.f, 0.f, 0.f};
#pragma unroll
        for (int kc = 0; kc < 4; kc++)
          c = wmma16(aH2[kc], load_frag(wp, 40 + nt * 4 + kc, ln), c);
        cs[nt] = c;
      }
#pragma unroll
      for (int nt = 0; nt < 3; nt++) {
        int col = nt * 16 + m;
        if (col < 36) {
          float bbv = bb3[col];
#pragma unroll
          for (int r = 0; r < 8; r++) outbm[wid][r + 8 * hf][col] = cs[nt][r] + bbv;
        }
      }
    }
    __syncthreads();
    if (hf == 0) {
#pragma unroll
      for (int i = 0; i < 6; i++) {
        float sacc = 0.f;
#pragma unroll
        for (int j = 0; j < 6; j++) sacc += outbm[wid][m][i * 6 + j] * P[j];
        Bsum[i] += wk * sacc;
      }
    }
    __syncthreads();

    // ---------- V-MLP forward (two-phase) ----------
    {
      float bb[8];
#pragma unroll
      for (int nt = 0; nt < 8; nt++) bb[nt] = vb1[nt * 16 + m];
      v8f cs[8];
#pragma unroll
      for (int nt = 0; nt < 8; nt++) {
        v8f c = {0.f, 0.f, 0.f, 0.f, 0.f, 0.f, 0.f, 0.f};
        cs[nt] = wmma16(aZv, load_frag(wp, 52 + nt, ln), c);
      }
#pragma unroll
      for (int nt = 0; nt < 8; nt++) {
        int col = nt * 16 + m;
#pragma unroll
        for (int r = 0; r < 8; r++)
          act1[wid][r + 8 * hf][col] = (_Float16)fast_tanh(cs[nt][r] + bb[nt]);
      }
    }
    __syncthreads();
    dense128_tanh(act1[wid], act2[wid], wp, 60, vb2, m, hf, ln);
    __syncthreads();
    {  // V scalar + backward
      v16h aH2[4];
      build_afrag(act2[wid], m, hf, aH2);
      v8f cV = {0.f, 0.f, 0.f, 0.f, 0.f, 0.f, 0.f, 0.f};
#pragma unroll
      for (int kc = 0; kc < 4; kc++)
        cV = wmma16(aH2[kc], load_frag(wp, 92 + kc, ln), cV);
      if (m == 0) {
#pragma unroll
        for (int r = 0; r < 8; r++) vbuf[wid][r + 8 * hf] = cV[r] + vb3[0];
      }
      // G2 = (1 - h2^2) * w3  (A fragments; w3 pre-packed in A-layout)
      v16h aG2[4];
#pragma unroll
      for (int kc = 0; kc < 4; kc++) {
        v16h w3f = load_frag(wp, 132 + kc, ln);
#pragma unroll
        for (int e = 0; e < 16; e++) {
          int K = kc * 32 + 16 * hf + e;
          float h2 = (float)act2[wid][m][K];
          aG2[kc][e] = (_Float16)((1.f - h2 * h2) * (float)w3f[e]);
        }
      }
      __syncthreads();
      // G1 = (1 - h1^2) * (G2 @ W2^T)  -> act2 (two-phase)
      {
        v8f cs[8];
#pragma unroll
        for (int nt = 0; nt < 8; nt++) {
          v8f c = {0.f, 0.f, 0.f, 0.f, 0.f, 0.f, 0.f, 0.f};
#pragma unroll
          for (int kc = 0; kc < 4; kc++)
            c = wmma16(aG2[kc], load_frag(wp, 96 + nt * 4 + kc, ln), c);
          cs[nt] = c;
        }
#pragma unroll
        for (int nt = 0; nt < 8; nt++) {
          int col = nt * 16 + m;
#pragma unroll
          for (int r = 0; r < 8; r++) {
            int row = r + 8 * hf;
            float h1 = (float)act1[wid][row][col];
            act2[wid][row][col] = (_Float16)((1.f - h1 * h1) * cs[nt][r]);
          }
        }
      }
      __syncthreads();
      // gV = G1 @ W1^T (cols 0..5)
      v16h aG1[4];
      build_afrag(act2[wid], m, hf, aG1);
      v8f cG = {0.f, 0.f, 0.f, 0.f, 0.f, 0.f, 0.f, 0.f};
#pragma unroll
      for (int kc = 0; kc < 4; kc++)
        cG = wmma16(aG1[kc], load_frag(wp, 128 + kc, ln), cG);
      if (m < 6) {
#pragma unroll
        for (int r = 0; r < 8; r++) gvbuf[wid][r + 8 * hf][m] = cG[r];
      }
    }
    __syncthreads();

    // ---------- per-sample chart finalize ----------
    if (hf == 0) {
      float Vk = vbuf[wid][m];
      float u[6];
#pragma unroll
      for (int i = 0; i < 6; i++) u[i] = wk * gvbuf[wid][m][i] + dwq[i] * Vk;
      float Kq[6][6];
      k2inv(qk, Kq);
#pragma unroll
      for (int i = 0; i < 6; i++) {
        float sdv = 0.f;
#pragma unroll
        for (int j = 0; j < 6; j++) sdv += Kq[j][i] * u[j];  // K^{-T} u
        dVa[i] += sdv;
      }
      if (k == ci) {
#pragma unroll
        for (int i = 0; i < 6; i++) {
          float sq = 0.f;
#pragma unroll
          for (int j = 0; j < 6; j++) sq += Kq[i][j] * T[j];  // K^{-1} T
          dqi[i] = sq;
        }
      }
    }
    __syncthreads();
  }

  // ================= epilogue =================
  if (hf == 0) {
    float adTP[6];
    adTP[0] = -(T[1] * P[2] - T[2] * P[1]) - (T[4] * P[5] - T[5] * P[4]);
    adTP[1] = -(T[2] * P[0] - T[0] * P[2]) - (T[5] * P[3] - T[3] * P[5]);
    adTP[2] = -(T[0] * P[1] - T[1] * P[0]) - (T[3] * P[4] - T[4] * P[3]);
    adTP[3] = -(T[1] * P[5] - T[2] * P[4]);
    adTP[4] = -(T[2] * P[3] - T[0] * P[5]);
    adTP[5] = -(T[0] * P[4] - T[1] * P[3]);
#pragma unroll
    for (int i = 0; i < 6; i++) {
      float Wf = -Bsum[i] - dVa[i];
      out[s * 13 + i]     = dqi[i];
      out[s * 13 + 6 + i] = adTP[i] + Wf;
    }
    out[s * 13 + 12] = 0.f;
  }
}

extern "C" void kernel_launch(void* const* d_in, const int* in_sizes, int n_in,
                              void* d_out, int out_size, void* d_ws, size_t ws_size,
                              hipStream_t stream) {
  (void)in_sizes; (void)n_in; (void)ws_size; (void)out_size;
  const float* x      = (const float*)d_in[1];
  const int*   ci     = (const int*)d_in[2];
  const float* I_diag = (const float*)d_in[3];
  const float* Vw1 = (const float*)d_in[4];
  const float* Vb1 = (const float*)d_in[5];
  const float* Vw2 = (const float*)d_in[6];
  const float* Vb2 = (const float*)d_in[7];
  const float* Vw3 = (const float*)d_in[8];
  const float* Vb3 = (const float*)d_in[9];
  const float* Bw1 = (const float*)d_in[10];
  const float* Bb1 = (const float*)d_in[11];
  const float* Bw2 = (const float*)d_in[12];
  const float* Bb2 = (const float*)d_in[13];
  const float* Bw3 = (const float*)d_in[14];
  const float* Bb3 = (const float*)d_in[15];
  float* out = (float*)d_out;
  _Float16* wpack = (_Float16*)d_ws;

  // pass 1: pack all WMMA B-fragments to f16 fragment-major layout in d_ws
  {
    int total = 4 * CHART_ELEMS;
    dim3 g((total + 255) / 256), b(256);
    hipLaunchKernelGGL(pack_weights, g, b, 0, stream, Vw1, Vw2, Vw3, Bw1, Bw2, Bw3, wpack);
  }
  // pass 2: main fused kernel
  {
    dim3 grid(NB / (16 * WPB));
    dim3 block(32 * WPB);
    hipLaunchKernelGGL(se3_dyn_kernel, grid, block, 0, stream,
                       x, ci, I_diag, Vb1, Vb2, Vb3, Bb1, Bb2, Bb3, wpack, out);
  }
}
